// MultiHeadAttention_65249143161356
// MI455X (gfx1250) — matrix operand order repaired and checked by reading, not compiled, not run on hardware
//
#include <hip/hip_runtime.h>
#include <hip/hip_bf16.h>

// ---------------------------------------------------------------------------
// Causal multi-head attention, bf16 WMMA pipeline for gfx1250 (MI455X).
//   B=8, S=1024, E=1024, H=16, HS=64
// Stages:
//   0) cast f32->bf16 (x, Wproj) and cast-transpose QKV weights to [H][HS][E]
//   1) QKV projection GEMM; V stored transposed [B,H,HS,S]
//   2) fused causal flash attention (online softmax; V loads issued ahead of
//      softmax; P transposed via bf16 LDS tile + s_wait_dscnt)
//   3) output projection GEMM (+bias), f32 output
// All matrix math uses v_wmma_f32_16x16x32_bf16; every B fragment is a
// K-contiguous 32-byte vector load.
// ---------------------------------------------------------------------------

constexpr int B_  = 8;
constexpr int S_  = 1024;
constexpr int E_  = 1024;
constexpr int H_  = 16;
constexpr int HS_ = 64;

typedef __attribute__((ext_vector_type(16))) __bf16 v16bf;
typedef __attribute__((ext_vector_type(8)))  __bf16 v8bf;
typedef __attribute__((ext_vector_type(4)))  __bf16 v4bf;
typedef __attribute__((ext_vector_type(8)))  float  v8f;
typedef __attribute__((ext_vector_type(4)))  float  v4f;

__device__ __forceinline__ v8f wmma_bf16(v16bf a, v16bf b, v8f c) {
  // D = A(16x32) * B(32x16) + C(16x16 f32)
  return __builtin_amdgcn_wmma_f32_16x16x32_bf16(
      /*neg_a=*/false, a, /*neg_b=*/false, b,
      /*c_mod=*/(short)0, c, /*reuse_a=*/false, /*reuse_b=*/false);
}

// A fragment (16x32 bf16, row-major source). base = row 0 of the 16-row tile,
// ld = row stride (elements), k0 = K offset (multiple of 32).
// Contiguous-K order (same as B loads): lane L holds row M=L%16; element i holds
//   K = k0 + 16*(L/16) + i   -> two 16-byte loads.
__device__ __forceinline__ v16bf load_a_rowmajor(const __bf16* base, int ld,
                                                 int k0, int lane) {
  const int hh = lane >> 4;
  const __bf16* p = base + (long)(lane & 15) * ld + k0 + hh * 16;
  v8bf lo = *(const v8bf*)p;
  v8bf hi = *(const v8bf*)(p + 8);
  return __builtin_shufflevector(lo, hi, 0, 1, 2, 3, 4, 5, 6, 7,
                                 8, 9, 10, 11, 12, 13, 14, 15);
}

// B fragment (32x16 bf16) where B[k][n] = base[n*row_stride + k]
// (K contiguous). Lane L holds col N=L%16; element i holds K = 16*(L/16)+i
// -> one 32-byte v16bf load per lane.
__device__ __forceinline__ v16bf load_b_kcontig(const __bf16* base,
                                                long row_stride, int lane) {
  const __bf16* p = base + (long)(lane & 15) * row_stride + ((lane >> 4) << 4);
  return *(const v16bf*)p;
}

// ---------------------------------------------------------------------------
// Stage 0a: vectorized f32 -> bf16 cast (n multiple of 4)
// ---------------------------------------------------------------------------
__global__ void k_cast4(const float* __restrict__ in,
                        __bf16* __restrict__ out, int n4) {
  int i = blockIdx.x * blockDim.x + threadIdx.x;
  if (i < n4) {
    v4f v = ((const v4f*)in)[i];
    ((v4bf*)out)[i] = __builtin_convertvector(v, v4bf);
  }
}

// ---------------------------------------------------------------------------
// Stage 0b: cast + transpose one QKV weight matrix [H,E,HS] -> [H,HS,E]
// ---------------------------------------------------------------------------
__global__ void k_cast_transpose(const float* __restrict__ in,
                                 __bf16* __restrict__ out, int n) {
  int i = blockIdx.x * blockDim.x + threadIdx.x;  // over H*E*HS
  if (i < n) {
    int h = i / (E_ * HS_);
    int rem = i % (E_ * HS_);
    int e = rem / HS_, d = rem % HS_;
    out[((long)h * HS_ + d) * E_ + e] = (__bf16)in[i];
  }
}

// ---------------------------------------------------------------------------
// Stage 1: QKV projection. One wave -> 16 rows x 64 cols (= one head's full
// HS for one of q/k/v). Weights pre-transposed: Wt[mat][h][d][e].
//   q,k stored [B,H,S,HS]; v stored transposed [B,H,HS,S].
// ---------------------------------------------------------------------------
__global__ void k_qkv(const __bf16* __restrict__ xb,   // [B*S, E]
                      const __bf16* __restrict__ wt3,  // [3][H][HS][E]
                      __bf16* __restrict__ qkv) {
  const int wave = (blockIdx.x * blockDim.x + threadIdx.x) >> 5;
  const int lane = threadIdx.x & 31;
  const int CG = 3 * H_;                 // 48 column groups of width HS
  const int mT = wave / CG;              // 0..511
  const int cg = wave % CG;
  const int mat = cg / H_, h = cg % H_;
  const long nQKV = (long)B_ * H_ * S_ * HS_;

  const __bf16* wbase = wt3 + (long)(mat * H_ + h) * HS_ * E_;
  const __bf16* xrow = xb + (long)(mT * 16) * E_;

  v8f acc[4] = {};
  for (int k0 = 0; k0 < E_; k0 += 32) {
    v16bf a = load_a_rowmajor(xrow, E_, k0, lane);
#pragma unroll
    for (int f = 0; f < 4; ++f) {
      // B[k][n] = Wt[h][16f+n][k0+k]  (K contiguous, stride E per column)
      v16bf b = load_b_kcontig(wbase + (long)(16 * f) * E_ + k0, E_, lane);
      acc[f] = wmma_bf16(a, b, acc[f]);
    }
  }

  const int hh = lane >> 4, col = lane & 15;
  __bf16* outm = qkv + (long)mat * nQKV;
#pragma unroll
  for (int f = 0; f < 4; ++f) {
    const int d = 16 * f + col;
#pragma unroll
    for (int r = 0; r < 8; ++r) {
      const int m = mT * 16 + r + 8 * hh;  // global row in [B*S)
      const int bb = m >> 10, s = m & (S_ - 1);
      if (mat < 2)
        outm[(((long)bb * H_ + h) * S_ + s) * HS_ + d] = (__bf16)acc[f][r];
      else  // V transposed: [B,H,HS,S]
        outm[(((long)bb * H_ + h) * HS_ + d) * S_ + s] = (__bf16)acc[f][r];
    }
  }
}

// ---------------------------------------------------------------------------
// Stage 2: fused causal flash attention per (b,h). One wave per 16-query
// tile; 32-key tiles; online softmax; P transposed through bf16 LDS tile.
// V fragments for the current tile are issued before the softmax VALU block
// so the loads overlap the transcendental work.
// ---------------------------------------------------------------------------
__global__ void k_flash(const __bf16* __restrict__ qm,  // [B,H,S,HS]
                        const __bf16* __restrict__ km,  // [B,H,S,HS]
                        const __bf16* __restrict__ vt,  // [B,H,HS,S]
                        __bf16* __restrict__ attn) {    // [B,S,E]
  __shared__ __bf16 Pb[8][16][32];  // per-wave 16x32 P tile (bf16)

  const int wib = threadIdx.x >> 5;
  const int lane = threadIdx.x & 31;
  const int wave = blockIdx.x * (blockDim.x >> 5) + wib;
  const int qTiles = S_ / 16;  // 64
  const int bh = wave / qTiles;
  const int q0 = (wave % qTiles) * 16;

  const __bf16* Q = qm + (long)bh * S_ * HS_;
  const __bf16* K = km + (long)bh * S_ * HS_;
  const __bf16* V = vt + (long)bh * HS_ * S_;  // transposed [HS][S]

  const int hh = lane >> 4, col = lane & 15;

  // Q A-fragments for the two 32-wide d slices, loaded once.
  v16bf aq0 = load_a_rowmajor(Q + (long)q0 * HS_, HS_, 0, lane);
  v16bf aq1 = load_a_rowmajor(Q + (long)q0 * HS_, HS_, 32, lane);

  v8f O0 = {}, O1 = {}, O2 = {}, O3 = {};
  float run_m[8], lsum[8];
#pragma unroll
  for (int r = 0; r < 8; ++r) { run_m[r] = -3.0e38f; lsum[r] = 0.0f; }

  for (int j0 = 0; j0 <= q0 + 15; j0 += 32) {
    // scores: 16x32 tile as two 16x16 fragments; K rows are d-contiguous
    v8f sc0 = {}, sc1 = {};
    {
      const __bf16* Kb0 = K + (long)j0 * HS_;
      const __bf16* Kb1 = K + (long)(j0 + 16) * HS_;
      sc0 = wmma_bf16(aq0, load_b_kcontig(Kb0, HS_, lane), sc0);
      sc0 = wmma_bf16(aq1, load_b_kcontig(Kb0 + 32, HS_, lane), sc0);
      sc1 = wmma_bf16(aq0, load_b_kcontig(Kb1, HS_, lane), sc1);
      sc1 = wmma_bf16(aq1, load_b_kcontig(Kb1 + 32, HS_, lane), sc1);
    }

    // V fragments for THIS tile: issue now so they are in flight during
    // the transcendental-heavy softmax below.
    v16bf vb0 = load_b_kcontig(V + 0L * S_ + j0,  S_, lane);
    v16bf vb1 = load_b_kcontig(V + 16L * S_ + j0, S_, lane);
    v16bf vb2 = load_b_kcontig(V + 32L * S_ + j0, S_, lane);
    v16bf vb3 = load_b_kcontig(V + 48L * S_ + j0, S_, lane);

    // online softmax (row = q0 + r + 8*hh, cols = j0+col, j0+16+col)
#pragma unroll
    for (int r = 0; r < 8; ++r) {
      const int qr = q0 + r + 8 * hh;
      float s0 = sc0[r] * 0.125f;  // HS^-0.5
      float s1 = sc1[r] * 0.125f;
      if (j0 + col > qr)      s0 = -1.0e30f;
      if (j0 + 16 + col > qr) s1 = -1.0e30f;

      float mx = fmaxf(s0, s1);
      mx = fmaxf(mx, __shfl_xor(mx, 1));
      mx = fmaxf(mx, __shfl_xor(mx, 2));
      mx = fmaxf(mx, __shfl_xor(mx, 4));
      mx = fmaxf(mx, __shfl_xor(mx, 8));  // row max within 16-lane half

      const float nm = fmaxf(run_m[r], mx);
      const float alpha = __expf(run_m[r] - nm);
      run_m[r] = nm;
      const float p0 = __expf(s0 - nm);
      const float p1 = __expf(s1 - nm);
      float ps = p0 + p1;
      ps += __shfl_xor(ps, 1);
      ps += __shfl_xor(ps, 2);
      ps += __shfl_xor(ps, 4);
      ps += __shfl_xor(ps, 8);
      lsum[r] = lsum[r] * alpha + ps;

      O0[r] *= alpha; O1[r] *= alpha; O2[r] *= alpha; O3[r] *= alpha;

      Pb[wib][r + 8 * hh][col]      = (__bf16)p0;
      Pb[wib][r + 8 * hh][16 + col] = (__bf16)p1;
    }

    // LDS writes -> reads within the wave: wait on DScnt (CDNA5 counter)
    asm volatile("s_wait_dscnt 0" ::: "memory");

    // reload P as an A fragment: row = lane&15, two contiguous 8-elem runs
    const __bf16* prow = &Pb[wib][col][0];
    v8bf plo = *(const v8bf*)(prow + hh * 16);
    v8bf phi = *(const v8bf*)(prow + 8 + hh * 16);
    v16bf ap = __builtin_shufflevector(plo, phi, 0, 1, 2, 3, 4, 5, 6, 7,
                                       8, 9, 10, 11, 12, 13, 14, 15);

    // O += P(16x32) x V(32x64)
    O0 = wmma_bf16(ap, vb0, O0);
    O1 = wmma_bf16(ap, vb1, O1);
    O2 = wmma_bf16(ap, vb2, O2);
    O3 = wmma_bf16(ap, vb3, O3);
  }

  // normalize and store concat-head layout [B][S][H*HS] as bf16
  const int bb = bh / H_, h = bh % H_;
#pragma unroll
  for (int r = 0; r < 8; ++r) {
    const float inv = 1.0f / lsum[r];
    const int s = q0 + r + 8 * hh;
    const long rowoff = ((long)bb * S_ + s) * E_ + (long)h * HS_;
    attn[rowoff + 0  + col] = (__bf16)(O0[r] * inv);
    attn[rowoff + 16 + col] = (__bf16)(O1[r] * inv);
    attn[rowoff + 32 + col] = (__bf16)(O2[r] * inv);
    attn[rowoff + 48 + col] = (__bf16)(O3[r] * inv);
  }
}

// ---------------------------------------------------------------------------
// Stage 3: output projection. One wave -> 16 rows x 64 cols.
//   y[m,n] = sum_e attn[m,e]*Wproj[n,e] + b[n]  (Wproj rows K-contiguous)
// ---------------------------------------------------------------------------
__global__ void k_proj(const __bf16* __restrict__ attn,  // [B*S, E]
                       const __bf16* __restrict__ wp,    // [E][E] (out,in)
                       const float* __restrict__ bias,
                       float* __restrict__ out) {
  const int wave = (blockIdx.x * blockDim.x + threadIdx.x) >> 5;
  const int lane = threadIdx.x & 31;
  const int NG = E_ / 64;  // 16 column groups
  const int mT = wave / NG, ng = wave % NG;
  const int n0 = ng * 64;

  const __bf16* arow = attn + (long)(mT * 16) * E_;
  const __bf16* wrow = wp + (long)n0 * E_;

  v8f acc[4] = {};
  for (int k0 = 0; k0 < E_; k0 += 32) {
    v16bf a = load_a_rowmajor(arow, E_, k0, lane);
#pragma unroll
    for (int f = 0; f < 4; ++f) {
      v16bf b = load_b_kcontig(wrow + (long)(16 * f) * E_ + k0, E_, lane);
      acc[f] = wmma_bf16(a, b, acc[f]);
    }
  }

  const int hh = lane >> 4, col = lane & 15;
#pragma unroll
  for (int f = 0; f < 4; ++f) {
    const int n = n0 + 16 * f + col;
    const float bn = bias[n];
#pragma unroll
    for (int r = 0; r < 8; ++r) {
      const int m = mT * 16 + r + 8 * hh;
      out[(long)m * E_ + n] = acc[f][r] + bn;
    }
  }
}

// ---------------------------------------------------------------------------
// Host launcher
// ---------------------------------------------------------------------------
extern "C" void kernel_launch(void* const* d_in, const int* in_sizes, int n_in,
                              void* d_out, int out_size, void* d_ws,
                              size_t ws_size, hipStream_t stream) {
  const float* x     = (const float*)d_in[0];   // [B,S,E]
  const float* Wq    = (const float*)d_in[1];   // [H,E,HS]
  const float* Wk    = (const float*)d_in[2];
  const float* Wv    = (const float*)d_in[3];
  const float* Wproj = (const float*)d_in[4];   // [E,E]
  const float* bproj = (const float*)d_in[5];   // [E]
  float* out = (float*)d_out;

  const long nX = (long)B_ * S_ * E_;          // 8,388,608
  const long nW = (long)H_ * E_ * HS_;         // 1,048,576
  const long nP = (long)E_ * E_;               // 1,048,576
  const long nQKV = (long)B_ * H_ * S_ * HS_;  // 8,388,608

  __bf16* wsb  = (__bf16*)d_ws;
  __bf16* xb   = wsb;             // nX
  __bf16* wt3  = xb + nX;         // 3*nW  ([3][H][HS][E], transposed)
  __bf16* wp   = wt3 + 3 * nW;    // nP
  __bf16* qkv  = wp + nP;         // 3*nQKV (q,k:[B,H,S,HS]; v:[B,H,HS,S])
  __bf16* attn = qkv + 3 * nQKV;  // nX    ([B][S][E])

  const int CB = 256;
  k_cast4<<<(int)(nX / 4 / CB), CB, 0, stream>>>(x, xb, (int)(nX / 4));
  k_cast_transpose<<<(int)((nW + CB - 1) / CB), CB, 0, stream>>>(Wq, wt3, (int)nW);
  k_cast_transpose<<<(int)((nW + CB - 1) / CB), CB, 0, stream>>>(Wk, wt3 + nW, (int)nW);
  k_cast_transpose<<<(int)((nW + CB - 1) / CB), CB, 0, stream>>>(Wv, wt3 + 2 * nW, (int)nW);
  k_cast4<<<(int)(nP / 4 / CB), CB, 0, stream>>>(Wproj, wp, (int)(nP / 4));

  // QKV projection: 512 row-tiles * 48 column groups = 24576 waves
  {
    const int waves = ((B_ * S_) / 16) * (3 * H_);
    k_qkv<<<waves / 8, 256, 0, stream>>>(xb, wt3, qkv);
  }

  // flash attention: (B*H) * (S/16) = 8192 waves
  {
    const int waves = (B_ * H_) * (S_ / 16);
    k_flash<<<waves / 8, 256, 0, stream>>>(qkv, qkv + nQKV, qkv + 2 * nQKV,
                                           attn);
  }

  // output projection: 512 row-tiles * 16 column groups = 8192 waves
  {
    const int waves = ((B_ * S_) / 16) * (E_ / 64);
    k_proj<<<waves / 8, 256, 0, stream>>>(attn, wp, bproj, out);
  }
}
